// SubgraphMamba_32074815767063
// MI455X (gfx1250) — compile-verified, matrix operand order checked by reading, not run on hardware
//
#include <hip/hip_runtime.h>
#include <math.h>

// ---------------- model constants ----------------
#define B_    32
#define L_    1024
#define DM_   512
#define DI_   1024
#define DS_   16
#define DTR_  32
#define NCLS_ 40
#define M_TOK (B_ * L_)          // 32768 tokens; always a multiple of BM

// ---------------- WMMA types ----------------
typedef __bf16 bf16_t;
typedef __attribute__((ext_vector_type(16))) __bf16 v16bf;
typedef __attribute__((ext_vector_type(8)))  __bf16 bf16x8;
typedef __attribute__((ext_vector_type(8)))  float  v8f;
typedef __attribute__((ext_vector_type(4)))  float  float4v;

union FragBF { v16bf v; unsigned int u[8]; };

// ---------------- tiled bf16 WMMA GEMM ----------------
// out[M,N](f32) = A[M,K](f32 row-major, optional row gather) * W[N,K]^T (f32) [+bias]
// EP: 0 = store, 1 = residual add into out, 2 = softplus(x+bias) store
// Requires: M % 128 == 0, K % 32 == 0, 16B-aligned pointers, lda/ldw % 4 == 0.
#define BM 128
#define BN 128
#define BK 32
#define LDSS 40   // padded LDS row stride (bf16): 80B -> 16B-aligned 8-elem stores

template <bool GATHER, int EP>
__global__ __launch_bounds__(256) void gemm_wmma_bf16(
    const float* __restrict__ A, int lda, const int* __restrict__ rowidx,
    const float* __restrict__ W, int ldw,
    const float* __restrict__ bias,
    float* __restrict__ out, int ldo,
    int N, int K)
{
  __shared__ bf16_t As[BM * LDSS];
  __shared__ bf16_t Ws[BN * LDSS];

  const int tid  = threadIdx.x;
  const int lane = tid & 31;
  const int wave = tid >> 5;
  const int lo   = lane & 15;     // M (A frag) / N (B frag) index within 16-tile
  const int hi   = lane >> 4;     // K-half select
  const int wm   = wave & 3;      // 4 waves tile M: 32 rows each
  const int wn   = wave >> 2;     // 2 waves tile N: 64 cols each
  const int m0   = blockIdx.y * BM;
  const int n0   = blockIdx.x * BN;

  // staging assignment: thread -> (row, 16-col half) of the 128x32 tile
  const int sr = tid >> 1;             // 0..127
  const int sk = (tid & 1) * 16;       // 0 or 16

  long arow = GATHER ? (long)rowidx[m0 + sr] : (long)(m0 + sr);
  const float* aptr = A + arow * (long)lda + sk;
  const bool wvalid = (n0 + sr) < N;
  const float* wptr = W + (long)(n0 + sr) * ldw + sk;

  v8f acc[2][4];
  const v8f vzero = {0.f, 0.f, 0.f, 0.f, 0.f, 0.f, 0.f, 0.f};
#pragma unroll
  for (int mt = 0; mt < 2; ++mt)
#pragma unroll
    for (int nt = 0; nt < 4; ++nt)
      acc[mt][nt] = vzero;

  const float4v fzero = {0.f, 0.f, 0.f, 0.f};
  float4v abuf[4], wbuf[4];

  // prologue load of k-tile 0 (4x b128 per operand per thread)
#pragma unroll
  for (int i = 0; i < 4; ++i) {
    abuf[i] = *(const float4v*)(aptr + 4 * i);
    wbuf[i] = wvalid ? *(const float4v*)(wptr + 4 * i) : fzero;
  }

  for (int k0 = 0; k0 < K; k0 += BK) {
    // ---- commit staged registers to LDS as packed bf16 (16B stores) ----
#pragma unroll
    for (int h = 0; h < 2; ++h) {
      bf16x8 pa, pw;
#pragma unroll
      for (int j = 0; j < 8; ++j) {
        pa[j] = (bf16_t)abuf[2 * h + (j >> 2)][j & 3];
        pw[j] = (bf16_t)wbuf[2 * h + (j >> 2)][j & 3];
      }
      *(bf16x8*)&As[sr * LDSS + sk + 8 * h] = pa;
      *(bf16x8*)&Ws[sr * LDSS + sk + 8 * h] = pw;
    }
    __syncthreads();

    // ---- overlap: issue next k-tile's global loads before the WMMAs ----
    if (k0 + BK < K) {
#pragma unroll
      for (int i = 0; i < 4; ++i) {
        abuf[i] = *(const float4v*)(aptr + k0 + BK + 4 * i);
        wbuf[i] = wvalid ? *(const float4v*)(wptr + k0 + BK + 4 * i) : fzero;
      }
      if (k0 + 2 * BK < K) {
        __builtin_prefetch(aptr + k0 + 2 * BK, 0, 0);   // global_prefetch_b8
        __builtin_prefetch(wptr + k0 + 2 * BK, 0, 0);
      }
    }

    // ---- load fragments per ISA 16-bit layout (K pairs per dword) ----
    FragBF a[2], b[4];
#pragma unroll
    for (int mt = 0; mt < 2; ++mt) {
      int row = wm * 32 + mt * 16 + lo;
#pragma unroll
      for (int i = 0; i < 8; ++i) {
        int g = i >> 2, j = i & 3;
        int k = g * 16 + hi * 8 + j * 2;
        a[mt].u[i] = *reinterpret_cast<const unsigned int*>(&As[row * LDSS + k]);
      }
    }
#pragma unroll
    for (int nt = 0; nt < 4; ++nt) {
      int row = wn * 64 + nt * 16 + lo;
#pragma unroll
      for (int i = 0; i < 8; ++i) {
        int g = i >> 2, j = i & 3;
        int k = g * 16 + hi * 8 + j * 2;
        b[nt].u[i] = *reinterpret_cast<const unsigned int*>(&Ws[row * LDSS + k]);
      }
    }

    // ---- 8x v_wmma_f32_16x16x32_bf16 per wave per k-step ----
#pragma unroll
    for (int mt = 0; mt < 2; ++mt)
#pragma unroll
      for (int nt = 0; nt < 4; ++nt)
        acc[mt][nt] = __builtin_amdgcn_wmma_f32_16x16x32_bf16(
            false, a[mt].v, false, b[nt].v, (short)0, acc[mt][nt], false, false);

    __syncthreads();
  }

  // ---- epilogue: D layout = lane lo -> N col, VGPR r + 8*hi -> M row ----
#pragma unroll
  for (int mt = 0; mt < 2; ++mt) {
#pragma unroll
    for (int nt = 0; nt < 4; ++nt) {
      int col = n0 + wn * 64 + nt * 16 + lo;
      if (col >= N) continue;
      int rbase = m0 + wm * 32 + mt * 16 + hi * 8;
      float bv = bias ? bias[col] : 0.f;
#pragma unroll
      for (int r = 0; r < 8; ++r) {
        long idx = (long)(rbase + r) * ldo + col;
        float v = acc[mt][nt][r] + bv;
        if (EP == 1)      out[idx] += v;
        else if (EP == 2) out[idx] = (v > 20.f) ? v : log1pf(expf(v));
        else              out[idx] = v;
      }
    }
  }
}

// ---------------- RMSNorm (one block per token row, d=512) ----------------
__global__ __launch_bounds__(256) void rmsnorm_kernel(
    const float* __restrict__ x, const float* __restrict__ scale,
    float* __restrict__ out)
{
  long row = blockIdx.x;
  const float* xr = x + row * DM_;
  int t = threadIdx.x;
  float v0 = xr[t], v1 = xr[t + 256];
  __shared__ float red[256];
  red[t] = v0 * v0 + v1 * v1;
  __syncthreads();
  for (int s = 128; s > 0; s >>= 1) {
    if (t < s) red[t] += red[t + s];
    __syncthreads();
  }
  float rms = sqrtf(red[0] / (float)DM_);
  float inv = 1.f / (rms + 1e-8f);
  out[row * DM_ + t]       = scale[t]       * v0 * inv;
  out[row * DM_ + t + 256] = scale[t + 256] * v1 * inv;
}

// ---------------- depthwise causal conv (DCONV=4) + SiLU ----------------
// xin lives in xz[..., :DI] (row stride 2*DI); output -> xc (B,L,DI)
__global__ __launch_bounds__(256) void conv_silu_kernel(
    const float* __restrict__ xz, const float* __restrict__ cw,
    const float* __restrict__ cb, float* __restrict__ xc)
{
  long idx = (long)blockIdx.x * 256 + threadIdx.x;   // over B*L*DI
  if (idx >= (long)M_TOK * DI_) return;
  int d   = (int)(idx % DI_);
  long bl = idx / DI_;        // b*L + l
  int l   = (int)(bl % L_);
  float w0 = cw[d * 4 + 0], w1 = cw[d * 4 + 1], w2 = cw[d * 4 + 2], w3 = cw[d * 4 + 3];
  const float* base = xz + bl * (2L * DI_) + d;
  float acc = cb[d] + w3 * base[0];
  if (l >= 1) acc += w2 * base[-1L * 2 * DI_];
  if (l >= 2) acc += w1 * base[-2L * 2 * DI_];
  if (l >= 3) acc += w0 * base[-3L * 2 * DI_];
  xc[idx] = acc / (1.f + expf(-acc));   // SiLU
}

// ---------------- selective scan + skip + gating ----------------
// h state (16) in registers; B_t/C_t double-buffered via LDS (1 barrier/step);
// writes gated y into xz[..., :DI] (z read from xz[..., DI:]).
__global__ __launch_bounds__(256) void scan_kernel(
    const float* __restrict__ delta, const float* __restrict__ xc,
    const float* __restrict__ xdbl, const float* __restrict__ A_log,
    const float* __restrict__ Dp, float* __restrict__ xz)
{
  int tid = threadIdx.x;
  int d = blockIdx.x * 256 + tid;   // channel
  int b = blockIdx.y;
  float a[DS_];
#pragma unroll
  for (int s = 0; s < DS_; ++s) a[s] = -expf(A_log[d * DS_ + s]);
  float Dv = Dp[d];
  float h[DS_];
#pragma unroll
  for (int s = 0; s < DS_; ++s) h[s] = 0.f;

  __shared__ float sB[2][DS_], sC[2][DS_];
  const long blbase = (long)b * L_;
  // preload timestep 0
  if (tid < 32) {
    float v = xdbl[blbase * 64 + DTR_ + tid];
    if (tid < DS_) sB[0][tid] = v;
    else           sC[0][tid - DS_] = v;
  }
  for (int l = 0; l < L_; ++l) {
    __syncthreads();
    int cur = l & 1;
    if (l + 1 < L_ && tid < 32) {       // stage next step into alternate buffer
      float v = xdbl[(blbase + l + 1) * 64 + DTR_ + tid];
      if (tid < DS_) sB[cur ^ 1][tid] = v;
      else           sC[cur ^ 1][tid - DS_] = v;
    }
    long bl = blbase + l;
    float dv = delta[bl * DI_ + d];
    float xv = xc[bl * DI_ + d];
    float dx = dv * xv;
    float y = 0.f;
#pragma unroll
    for (int s = 0; s < DS_; ++s) {
      float dA = expf(dv * a[s]);
      h[s] = h[s] * dA + dx * sB[cur][s];
      y += h[s] * sC[cur][s];
    }
    float z   = xz[bl * (2L * DI_) + DI_ + d];
    float sil = z / (1.f + expf(-z));
    xz[bl * (2L * DI_) + d] = (y + Dv * xv) * sil;
  }
}

// ---------------- masked mean pool ----------------
__global__ __launch_bounds__(256) void pool_kernel(
    const float* __restrict__ subg, const float* __restrict__ mask,
    float* __restrict__ pooled)
{
  int dm = blockIdx.x * 256 + threadIdx.x;  // < 512
  int b  = blockIdx.y;
  float s = 0.f, c = 0.f;
  for (int l = 0; l < L_; ++l) {
    float m = mask[b * L_ + l];
    s += subg[((long)b * L_ + l) * DM_ + dm] * m;
    c += m;
  }
  pooled[b * DM_ + dm] = s / (c + 1e-5f);
}

// ---------------- classifier head ----------------
__global__ __launch_bounds__(256) void fc2_kernel(
    const float* __restrict__ pooled, const float* __restrict__ w,
    const float* __restrict__ bias, float* __restrict__ out)
{
  int idx = blockIdx.x * 256 + threadIdx.x;
  if (idx >= B_ * NCLS_) return;
  int b = idx / NCLS_, c = idx % NCLS_;
  float acc = bias[c];
  for (int k = 0; k < DM_; ++k) acc += pooled[b * DM_ + k] * w[c * DM_ + k];
  out[idx] = acc;
}

// ---------------- host-side orchestration ----------------
extern "C" void kernel_launch(void* const* d_in, const int* in_sizes, int n_in,
                              void* d_out, int out_size, void* d_ws, size_t ws_size,
                              hipStream_t stream) {
  const int*   seq        = (const int*)  d_in[0];
  const float* inclusion  = (const float*)d_in[1];
  const float* embeddings = (const float*)d_in[2];
  const float* fc1_w      = (const float*)d_in[3];
  const float* fc1_b      = (const float*)d_in[4];
  const float* fc2_w      = (const float*)d_in[5];
  const float* fc2_b      = (const float*)d_in[6];
  const float* norm_scale = (const float*)d_in[7];
  const float* in_proj_w  = (const float*)d_in[8];
  const float* conv_w     = (const float*)d_in[9];
  const float* conv_b     = (const float*)d_in[10];
  const float* x_proj_w   = (const float*)d_in[11];
  const float* dt_proj_w  = (const float*)d_in[12];
  const float* dt_proj_b  = (const float*)d_in[13];
  const float* A_log      = (const float*)d_in[14];
  const float* Dvec       = (const float*)d_in[15];
  const float* out_proj_w = (const float*)d_in[16];

  // workspace partition (fp32)
  float* subg   = (float*)d_ws;                         // 32768 x 512
  float* zi     = subg  + (long)M_TOK * DM_;            // 32768 x 512
  float* xz     = zi    + (long)M_TOK * DM_;            // 32768 x 2048
  float* xc     = xz    + (long)M_TOK * 2 * DI_;        // 32768 x 1024
  float* xdbl   = xc    + (long)M_TOK * DI_;            // 32768 x 64
  float* delta  = xdbl  + (long)M_TOK * 64;             // 32768 x 1024
  float* pooled = delta + (long)M_TOK * DI_;            // 32 x 512

  dim3 blk(256);
  const int M = M_TOK;

  // fc1 with gather folded into A-row indexing: subg = emb[seq] @ fc1_w^T + b
  gemm_wmma_bf16<true, 0><<<dim3(DM_ / BN, M / BM), blk, 0, stream>>>(
      embeddings, 512, seq, fc1_w, 512, fc1_b, subg, DM_, DM_, 512);

  for (int ly = 0; ly < 2; ++ly) {
    // prenorm
    rmsnorm_kernel<<<M, blk, 0, stream>>>(subg, norm_scale + ly * DM_, zi);
    // in_proj: (M,512) -> (M,2048)
    gemm_wmma_bf16<false, 0><<<dim3(2 * DI_ / BN, M / BM), blk, 0, stream>>>(
        zi, DM_, nullptr, in_proj_w + (long)ly * 2 * DI_ * DM_, DM_,
        nullptr, xz, 2 * DI_, 2 * DI_, DM_);
    // depthwise causal conv + SiLU
    long nconv = (long)M * DI_;
    conv_silu_kernel<<<(unsigned)((nconv + 255) / 256), blk, 0, stream>>>(
        xz, conv_w + (long)ly * DI_ * 4, conv_b + (long)ly * DI_, xc);
    // x_proj: (M,1024) -> (M,64)
    gemm_wmma_bf16<false, 0><<<dim3(1, M / BM), blk, 0, stream>>>(
        xc, DI_, nullptr, x_proj_w + (long)ly * 64 * DI_, DI_,
        nullptr, xdbl, 64, 64, DI_);
    // dt_proj + softplus: (M,32) -> (M,1024)
    gemm_wmma_bf16<false, 2><<<dim3(DI_ / BN, M / BM), blk, 0, stream>>>(
        xdbl, 64, nullptr, dt_proj_w + (long)ly * DI_ * DTR_, DTR_,
        dt_proj_b + (long)ly * DI_, delta, DI_, DI_, DTR_);
    // selective scan + D*x skip + SiLU(z) gate -> xz[:, :DI]
    scan_kernel<<<dim3(DI_ / 256, B_), blk, 0, stream>>>(
        delta, xc, xdbl, A_log + (long)ly * DI_ * DS_, Dvec + (long)ly * DI_, xz);
    // out_proj + residual: subg += y @ opw^T
    gemm_wmma_bf16<false, 1><<<dim3(DM_ / BN, M / BM), blk, 0, stream>>>(
        xz, 2 * DI_, nullptr, out_proj_w + (long)ly * DM_ * DI_, DI_,
        nullptr, subg, DM_, DM_, DI_);
  }

  pool_kernel<<<dim3(DM_ / 256, B_), blk, 0, stream>>>(subg, inclusion, pooled);
  fc2_kernel<<<(B_ * NCLS_ + 255) / 256, blk, 0, stream>>>(
      pooled, fc2_w, fc2_b, (float*)d_out);
}
